// Electronic_embedding_88622355185706
// MI455X (gfx1250) — matrix operand order repaired and checked by reading, not compiled, not run on hardware
//
#include <hip/hip_runtime.h>
#include <hip/hip_bf16.h>

// ---------------------------------------------------------------------------
// SpookyNet electronic-embedding fused pipeline for MI455X (gfx1250, wave32)
//   q-GEMM eliminated algebraically; the two residual MLP GEMMs run on
//   v_wmma_f32_16x16x32_bf16 (bf16 in, f32 accumulate).
// ---------------------------------------------------------------------------

#define FDIM 256
#define LDA  264   // padded LDS row stride (bf16 elems): 528B -> 4-bank lane stride

typedef __attribute__((ext_vector_type(16))) __bf16 v16bf;
typedef __attribute__((ext_vector_type(8)))  __bf16 v8bf;
typedef __attribute__((ext_vector_type(8)))  float  v8f;

union FragBF { v16bf v; v8bf h[2]; };

__device__ __forceinline__ float siluf(float x)     { return x / (1.0f + expf(-x)); }
__device__ __forceinline__ float softplusf(float x) { return fmaxf(x, 0.0f) + log1pf(expf(-fabsf(x))); }

// ---------------------------------------------------------------------------
// Kernel 0: u = Wq^T Wk, c0 = Wk.bq, zero anorm, W1/W2 -> bf16
// ---------------------------------------------------------------------------
__global__ void ee_prep_kernel(const float* __restrict__ Wq, const float* __restrict__ bq,
                               const float* __restrict__ Wk,
                               const float* __restrict__ W1, const float* __restrict__ W2,
                               float* __restrict__ u, float* __restrict__ c0,
                               float* __restrict__ anorm,
                               __bf16* __restrict__ W1bf, __bf16* __restrict__ W2bf, int B) {
  int idx = blockIdx.x * blockDim.x + threadIdx.x;
  if (idx < FDIM * FDIM) { W1bf[idx] = (__bf16)W1[idx]; return; }
  idx -= FDIM * FDIM;
  if (idx < FDIM * FDIM) { W2bf[idx] = (__bf16)W2[idx]; return; }
  idx -= FDIM * FDIM;
  if (idx < B) { anorm[idx] = 0.0f; return; }
  idx -= B;
  if (idx < FDIM) {
    float s = 0.0f;
    for (int j = 0; j < FDIM; ++j) s += Wk[j] * Wq[j * FDIM + idx];
    u[idx] = s;
    if (idx == 0) {
      float c = 0.0f;
      for (int j = 0; j < FDIM; ++j) c += Wk[j] * bq[j];
      *c0 = c;
    }
  }
}

// ---------------------------------------------------------------------------
// Kernel 1: per row  t = u.x[n] + c0 ; a = softplus(|E[seg]| * t / 16)
//           a_out[n] = a ; atomicAdd(anorm[seg], a).  One wave32 per row.
// ---------------------------------------------------------------------------
__global__ void __launch_bounds__(256) ee_attn_kernel(
    const float* __restrict__ x, const float* __restrict__ E,
    const long long* __restrict__ seg,
    const float* __restrict__ u, const float* __restrict__ c0,
    float* __restrict__ a_out, float* __restrict__ anorm, int N) {
  const int lane = threadIdx.x & 31;
  const int wave = threadIdx.x >> 5;
  const int n = blockIdx.x * 8 + wave;
  if (n >= N) return;

  const float4* xr = (const float4*)(x + (size_t)n * FDIM);
  const float4* ur = (const float4*)u;
  float4 x0 = xr[lane * 2], x1 = xr[lane * 2 + 1];
  float4 u0 = ur[lane * 2], u1 = ur[lane * 2 + 1];
  float t = x0.x * u0.x + x0.y * u0.y + x0.z * u0.z + x0.w * u0.w
          + x1.x * u1.x + x1.y * u1.y + x1.z * u1.z + x1.w * u1.w;
#pragma unroll
  for (int off = 16; off > 0; off >>= 1) t += __shfl_xor(t, off, 32);

  if (lane == 0) {
    long long sg = seg[n];
    float e   = fabsf(E[sg]);
    float dot = e * (t + *c0) * 0.0625f;          // / sqrt(256)
    float a   = softplusf(dot);
    a_out[n]  = a;
    atomicAdd(anorm + sg, a);
  }
}

// ---------------------------------------------------------------------------
// Kernel 2: fused  y = s*Wv ;  h = silu(y)@W1^T+b1 ; h = silu(h)@W2^T+b2 ;
//           out = y + h.   32 rows / block, 8 waves, WMMA bf16->f32.
// ---------------------------------------------------------------------------
__global__ void __launch_bounds__(256) ee_mlp_kernel(
    const float* __restrict__ E, const long long* __restrict__ seg,
    const float* __restrict__ a_in, const float* __restrict__ anorm,
    const float* __restrict__ Wv, const float* __restrict__ b1,
    const float* __restrict__ b2,
    const __bf16* __restrict__ W1bf, const __bf16* __restrict__ W2bf,
    float* __restrict__ out) {
  __shared__ __align__(16) __bf16 ldsA[32 * LDA];
  __shared__ float lds_s[32];

  const int tid  = threadIdx.x;
  const int lane = tid & 31;
  const int wave = tid >> 5;
  const int half = lane >> 4;       // 0: lanes 0-15, 1: lanes 16-31
  const int l15  = lane & 15;
  const int nb   = blockIdx.x * 32; // first row of this block (N % 32 == 0)

  // per-row scalar s[n] = (a / (anorm+eps)) * |E[seg]|
  if (tid < 32) {
    int n = nb + tid;
    long long sg = seg[n];
    float w = a_in[n] / (anorm[sg] + 1e-8f);
    lds_s[tid] = w * fabsf(E[sg]);
  }
  __syncthreads();

  // A1[m][f] = silu(s[m] * Wv[f]) as bf16 in padded LDS
  for (int idx = tid; idx < 32 * FDIM; idx += 256) {
    int m = idx >> 8, j = idx & 255;
    ldsA[m * LDA + j] = (__bf16)siluf(lds_s[m] * Wv[j]);
  }
  __syncthreads();

  const int jb = wave * 32;          // this wave's 32 output columns
  v8f acc[2][2];
#pragma unroll
  for (int mt = 0; mt < 2; ++mt)
#pragma unroll
    for (int nt = 0; nt < 2; ++nt)
#pragma unroll
      for (int r = 0; r < 8; ++r) acc[mt][nt][r] = 0.0f;

  // ---------------- layer 1: h1 = A1 @ W1^T ----------------
#pragma unroll
  for (int ks = 0; ks < 8; ++ks) {
    const int kb = ks * 32;
    FragBF A0, A1, B0, B1;
    // A 16x32 bf16, contiguous k order used here: M = lane&15, K = e + (lane>=16 ? 16 : 0)
    const int f0 = kb + half * 16;
    A0.h[0] = *(const v8bf*)(&ldsA[(0  + l15) * LDA + f0]);
    A0.h[1] = *(const v8bf*)(&ldsA[(0  + l15) * LDA + f0 + 8]);
    A1.h[0] = *(const v8bf*)(&ldsA[(16 + l15) * LDA + f0]);
    A1.h[1] = *(const v8bf*)(&ldsA[(16 + l15) * LDA + f0 + 8]);
    // B 32x16 bf16, contiguous k order used here: N = lane&15, K = e + (lane>=16 ? 16 : 0)
    const int fb = kb + half * 16;
    B0.v = *(const v16bf*)(W1bf + (size_t)(jb      + l15) * FDIM + fb);
    B1.v = *(const v16bf*)(W1bf + (size_t)(jb + 16 + l15) * FDIM + fb);

    acc[0][0] = __builtin_amdgcn_wmma_f32_16x16x32_bf16(false, A0.v, false, B0.v, (short)0, acc[0][0], false, false);
    acc[0][1] = __builtin_amdgcn_wmma_f32_16x16x32_bf16(false, A0.v, false, B1.v, (short)0, acc[0][1], false, false);
    acc[1][0] = __builtin_amdgcn_wmma_f32_16x16x32_bf16(false, A1.v, false, B0.v, (short)0, acc[1][0], false, false);
    acc[1][1] = __builtin_amdgcn_wmma_f32_16x16x32_bf16(false, A1.v, false, B1.v, (short)0, acc[1][1], false, false);
  }
  __syncthreads();   // all reads of A1 done before overwrite

  // silu(h1 + b1) -> bf16 back into LDS
#pragma unroll
  for (int nt = 0; nt < 2; ++nt) {
    const int j = jb + nt * 16 + l15;
    const float bias = b1[j];
#pragma unroll
    for (int mt = 0; mt < 2; ++mt)
#pragma unroll
      for (int r = 0; r < 8; ++r) {
        const int m = mt * 16 + half * 8 + r;   // D layout: M = r (+8 for hi lanes)
        ldsA[m * LDA + j] = (__bf16)siluf(acc[mt][nt][r] + bias);
      }
  }
  __syncthreads();

  // ---------------- layer 2: h2 = silu(h1) @ W2^T ----------------
#pragma unroll
  for (int mt = 0; mt < 2; ++mt)
#pragma unroll
    for (int nt = 0; nt < 2; ++nt)
#pragma unroll
      for (int r = 0; r < 8; ++r) acc[mt][nt][r] = 0.0f;

#pragma unroll
  for (int ks = 0; ks < 8; ++ks) {
    const int kb = ks * 32;
    FragBF A0, A1, B0, B1;
    const int f0 = kb + half * 16;
    A0.h[0] = *(const v8bf*)(&ldsA[(0  + l15) * LDA + f0]);
    A0.h[1] = *(const v8bf*)(&ldsA[(0  + l15) * LDA + f0 + 8]);
    A1.h[0] = *(const v8bf*)(&ldsA[(16 + l15) * LDA + f0]);
    A1.h[1] = *(const v8bf*)(&ldsA[(16 + l15) * LDA + f0 + 8]);
    const int fb = kb + half * 16;
    B0.v = *(const v16bf*)(W2bf + (size_t)(jb      + l15) * FDIM + fb);
    B1.v = *(const v16bf*)(W2bf + (size_t)(jb + 16 + l15) * FDIM + fb);

    acc[0][0] = __builtin_amdgcn_wmma_f32_16x16x32_bf16(false, A0.v, false, B0.v, (short)0, acc[0][0], false, false);
    acc[0][1] = __builtin_amdgcn_wmma_f32_16x16x32_bf16(false, A0.v, false, B1.v, (short)0, acc[0][1], false, false);
    acc[1][0] = __builtin_amdgcn_wmma_f32_16x16x32_bf16(false, A1.v, false, B0.v, (short)0, acc[1][0], false, false);
    acc[1][1] = __builtin_amdgcn_wmma_f32_16x16x32_bf16(false, A1.v, false, B1.v, (short)0, acc[1][1], false, false);
  }

  // epilogue: out = y + h2 + b2 = s*Wv + acc + b2
#pragma unroll
  for (int nt = 0; nt < 2; ++nt) {
    const int j = jb + nt * 16 + l15;
    const float bias = b2[j];
    const float wv   = Wv[j];
#pragma unroll
    for (int mt = 0; mt < 2; ++mt)
#pragma unroll
      for (int r = 0; r < 8; ++r) {
        const int m = mt * 16 + half * 8 + r;
        const int n = nb + m;
        out[(size_t)n * FDIM + j] = lds_s[m] * wv + bias + acc[mt][nt][r];
      }
  }
}

// ---------------------------------------------------------------------------
extern "C" void kernel_launch(void* const* d_in, const int* in_sizes, int n_in,
                              void* d_out, int out_size, void* d_ws, size_t ws_size,
                              hipStream_t stream) {
  const float*     x   = (const float*)d_in[0];
  const float*     E   = (const float*)d_in[1];
  const long long* seg = (const long long*)d_in[2];
  // d_in[3] = num_batch (scalar) — B taken from in_sizes[1] (= E length)
  const float* Wq = (const float*)d_in[4];
  const float* bq = (const float*)d_in[5];
  const float* Wk = (const float*)d_in[6];
  const float* Wv = (const float*)d_in[7];
  const float* W1 = (const float*)d_in[8];
  const float* b1 = (const float*)d_in[9];
  const float* W2 = (const float*)d_in[10];
  const float* b2 = (const float*)d_in[11];
  float* out = (float*)d_out;

  const int N = in_sizes[2];   // 300000 (divisible by 32)
  const int B = in_sizes[1];   // 4096

  // workspace layout (floats):
  //   [0,256)              u
  //   [256]                c0
  //   [320, 320+B)         anorm
  //   [320+B, 320+B+N)     a
  //   then W1bf, W2bf (bf16, 32B-aligned by construction)
  float* ws    = (float*)d_ws;
  float* u     = ws;
  float* c0    = ws + 256;
  float* anorm = ws + 320;
  float* a_arr = ws + 320 + B;
  __bf16* W1bf = (__bf16*)(ws + 320 + B + N);
  __bf16* W2bf = W1bf + FDIM * FDIM;

  const int prep_items = 2 * FDIM * FDIM + B + FDIM;
  ee_prep_kernel<<<(prep_items + 255) / 256, 256, 0, stream>>>(
      Wq, bq, Wk, W1, W2, u, c0, anorm, W1bf, W2bf, B);

  ee_attn_kernel<<<(N + 7) / 8, 256, 0, stream>>>(
      x, E, seg, u, c0, a_arr, anorm, N);

  ee_mlp_kernel<<<N / 32, 256, 0, stream>>>(
      E, seg, a_arr, anorm, Wv, b1, b2, W1bf, W2bf, out);
}
